// Model_76897094468146
// MI455X (gfx1250) — compile-verified, matrix operand order checked by reading, not run on hardware
//
#include <hip/hip_runtime.h>
#include <math.h>

#define K_NB   8
#define IN_F   68
#define EO     32
#define HFEAT  132                      // 68 + 32 + 32
#define FEAT   264
#define EMB    32
#define NCLS   2
#define NNODE  768
#define NPAIR_G ((NNODE*(NNODE-1))/2)   // 294528
#define NPAIR_T (2*NPAIR_G)             // 589056
#define NTILE   (NPAIR_T/16)            // 36816
#define EPSV    1e-5f

typedef __attribute__((ext_vector_type(16))) __bf16        v16bf;
typedef __attribute__((ext_vector_type(8)))  float         v8f;
typedef __attribute__((ext_vector_type(8)))  unsigned int  v8u;

__device__ __forceinline__ float lrelu(float x){ return x >= 0.f ? x : 0.01f * x; }

__device__ __forceinline__ unsigned short f2bf(float f){
  unsigned int u = __float_as_uint(f);
  u += 0x7fffu + ((u >> 16) & 1u);      // round-to-nearest-even
  return (unsigned short)(u >> 16);
}

// ---------------------------------------------------------------- KNN (top-8 L1)
__global__ void knn_kernel(const float* __restrict__ feat0,
                           const float* __restrict__ feat1,
                           int* __restrict__ idxbuf){
  __shared__ float c[NNODE][4];
  int g = blockIdx.x / 3, part = blockIdx.x % 3;
  const float* feat = g ? feat1 : feat0;
  for (int t = threadIdx.x; t < NNODE*4; t += blockDim.x){
    int j = t >> 2, d = t & 3;
    c[j][d] = feat[j*IN_F + d];
  }
  __syncthreads();
  int i = part*256 + (int)threadIdx.x;
  float c0=c[i][0], c1=c[i][1], c2=c[i][2], c3=c[i][3];
  float bd[K_NB]; int bi[K_NB];
  #pragma unroll
  for (int t=0;t<K_NB;t++){ bd[t]=3.4e38f; bi[t]=0; }
  for (int j=0;j<NNODE;j++){
    float d = fabsf(c0-c[j][0]) + fabsf(c1-c[j][1]) + fabsf(c2-c[j][2]) + fabsf(c3-c[j][3]);
    if (d < bd[K_NB-1]){                 // strict '<' => stable (earlier index wins ties)
      bd[K_NB-1]=d; bi[K_NB-1]=j;
      #pragma unroll
      for (int t=K_NB-1;t>0;--t){
        if (bd[t] < bd[t-1]){
          float td=bd[t]; bd[t]=bd[t-1]; bd[t-1]=td;
          int   ti=bi[t]; bi[t]=bi[t-1]; bi[t-1]=ti;
        }
      }
    }
  }
  #pragma unroll
  for (int t=0;t<K_NB;t++) idxbuf[(g*NNODE + i)*K_NB + t] = bi[t];
}

// ---------------------------------------------------------- EdgeConv (wave/node)
template<int F>
__global__ void edgeconv_kernel(const float* __restrict__ in0,
                                const float* __restrict__ in1,
                                const float* __restrict__ w1, const float* __restrict__ b1,
                                const float* __restrict__ w2, const float* __restrict__ b2,
                                const int* __restrict__ idxbuf,
                                float* __restrict__ outbuf){
  int lane  = threadIdx.x & 31;
  int gnode = blockIdx.x * (blockDim.x >> 5) + (threadIdx.x >> 5);
  int g = gnode / NNODE, i = gnode % NNODE;
  const float* in = g ? in1 : in0;
  float acc = b1[lane];
  const int* nb = idxbuf + (g*NNODE + i)*K_NB;
  for (int k=0;k<K_NB;k++){
    const float* row = in + nb[k]*F;
    for (int f=0; f<F; f++)
      acc += row[f] * w1[(f*K_NB + k)*EO + lane];  // v[f*K+k] layout
  }
  float t = lrelu(acc);
  float acc2 = b2[lane];
  for (int o1=0;o1<EO;o1++)
    acc2 += __shfl(t, o1) * w2[o1*EO + lane];
  outbuf[gnode*EO + lane] = lrelu(acc2);
}

// ------------------------------------------------------------------ x[g,i,r]
__device__ __forceinline__ float fetchx(int g, int i, int r,
    const float* f0, const float* f1,
    const float* e1, const float* e2){
  if (r < IN_F)      return (g ? f1 : f0)[i*IN_F + r];
  if (r < IN_F+EO)   return e1[(g*NNODE+i)*EO + (r-IN_F)];
  return e2[(g*NNODE+i)*EO + (r-IN_F-EO)];
}

// -------------------------------------------------- exact pair-batch BN stats
__global__ void stats_kernel(const float* __restrict__ f0, const float* __restrict__ f1,
                             const float* __restrict__ e1, const float* __restrict__ e2,
                             const float* __restrict__ bn_g, const float* __restrict__ bn_b,
                             float* __restrict__ scale, float* __restrict__ shift){
  int f = threadIdx.x;
  if (f >= FEAT) return;
  int first = (f < HFEAT);
  int r = first ? f : f - HFEAT;
  double s1 = 0.0, s2 = 0.0;
  for (int g=0; g<2; g++)
    for (int i=0;i<NNODE;i++){
      double x = (double)fetchx(g,i,r,f0,f1,e1,e2);
      double w = first ? (double)(NNODE-1-i) : (double)i;  // multiplicity in pair batch
      s1 += w*x; s2 += w*x*x;
    }
  double M  = (double)NPAIR_T;
  double mu = s1 / M;
  double var = s2 / M - mu*mu;
  float sc = bn_g[f] * rsqrtf((float)var + EPSV);
  scale[f] = sc;
  shift[f] = bn_b[f] - (float)mu * sc;
}

// ------------------- fold BN shift into lin1 bias; pack W2/W3 into WMMA B layout
__global__ void fold_kernel(const float* __restrict__ shift,
                            const float* __restrict__ lin1_w, const float* __restrict__ lin1_b,
                            const float* __restrict__ lin2_w, const float* __restrict__ lin3_w,
                            float* __restrict__ biasp,
                            unsigned int* __restrict__ w2pack,
                            unsigned int* __restrict__ w3pack){
  int o = threadIdx.x;
  if (o >= 32) return;
  float b = lin1_b[o];
  for (int r=0;r<FEAT;r++) b += shift[r] * lin1_w[r*EMB + o];
  biasp[o] = b;

  int col = o & 15, kh = o >> 4;
  // per-lane B operand: 16 bf16 = 8 dwords; t -> k map mirrors the 16-bit A layout
  for (int cb=0; cb<2; cb++){
    for (int d=0; d<8; d++){
      int t0 = 2*d, t1 = 2*d+1;
      int k0 = (t0<8) ? kh*8 + t0 : 16 + kh*8 + (t0-8);
      int k1 = (t1<8) ? kh*8 + t1 : 16 + kh*8 + (t1-8);
      int n  = cb*16 + col;
      unsigned int lo = f2bf(lin2_w[k0*EMB + n]);
      unsigned int hi = f2bf(lin2_w[k1*EMB + n]);
      w2pack[(cb*32 + o)*8 + d] = lo | (hi << 16);
    }
  }
  for (int d=0; d<8; d++){
    int t0 = 2*d, t1 = 2*d+1;
    int k0 = (t0<8) ? kh*8 + t0 : 16 + kh*8 + (t0-8);
    int k1 = (t1<8) ? kh*8 + t1 : 16 + kh*8 + (t1-8);
    unsigned int lo = (col < NCLS) ? (unsigned int)f2bf(lin3_w[k0*NCLS + col]) : 0u;
    unsigned int hi = (col < NCLS) ? (unsigned int)f2bf(lin3_w[k1*NCLS + col]) : 0u;
    w3pack[o*8 + d] = lo | (hi << 16);
  }
}

// -------------------------------- per-node lin1 half projections A[i], B[j]
__global__ void nodeproj_kernel(const float* __restrict__ f0, const float* __restrict__ f1,
                                const float* __restrict__ e1, const float* __restrict__ e2,
                                const float* __restrict__ scale,
                                const float* __restrict__ lin1_w,
                                float* __restrict__ Aout, float* __restrict__ Bout){
  int lane  = threadIdx.x & 31;
  int gnode = blockIdx.x * (blockDim.x >> 5) + (threadIdx.x >> 5);
  int g = gnode / NNODE, i = gnode % NNODE;
  float a = 0.f, b = 0.f;
  for (int r=0;r<HFEAT;r++){
    float x = fetchx(g,i,r,f0,f1,e1,e2);
    a += x * scale[r]         * lin1_w[ r        *EMB + lane];
    b += x * scale[HFEAT + r] * lin1_w[(HFEAT+r) *EMB + lane];
  }
  Aout[gnode*EMB + lane] = a;
  Bout[gnode*EMB + lane] = b;
}

// ------------------------------------------- pair MLP: the WMMA hot loop
__global__ void __launch_bounds__(256)
pair_kernel(const float* __restrict__ Aarr, const float* __restrict__ Barr,
            const float* __restrict__ biasp,
            const unsigned int* __restrict__ w2pack,
            const unsigned int* __restrict__ w3pack,
            float* __restrict__ out){
  __shared__ unsigned short hbuf[8][16][32];   // per-wave 16x32 bf16 tile
  int lane = threadIdx.x & 31;
  int w    = threadIdx.x >> 5;
  int tile = blockIdx.x * 8 + w;               // exactly NTILE waves launched
  int col  = lane & 15, kh = lane >> 4;

  // loop-invariant WMMA B operands (packed by fold_kernel)
  v8u w2u0 = {}, w2u1 = {}, w3u = {};
  #pragma unroll
  for (int d=0; d<8; d++){
    w2u0[d] = w2pack[(0*32 + lane)*8 + d];
    w2u1[d] = w2pack[(1*32 + lane)*8 + d];
    w3u[d]  = w3pack[lane*8 + d];
  }
  v16bf wb0 = __builtin_bit_cast(v16bf, w2u0);
  v16bf wb1 = __builtin_bit_cast(v16bf, w2u1);
  v16bf wb3 = __builtin_bit_cast(v16bf, w3u);
  float biasv = biasp[lane];

  int p0 = tile * 16;
  int g  = (p0 >= NPAIR_G) ? 1 : 0;            // NPAIR_G % 16 == 0: no straddle
  int q  = p0 - g*NPAIR_G + col;               // pair index for m = col
  // invert s(i) = i*(n-1) - i*(i-1)/2 <= q
  double disc = (double)(2*NNODE-1)*(2*NNODE-1) - 8.0*(double)q;
  int ii = (int)(((double)(2*NNODE-1) - sqrt(disc)) * 0.5);
  if (ii < 0) ii = 0; if (ii > NNODE-2) ii = NNODE-2;
  while (ii > 0 && (ii*(NNODE-1) - (ii*(ii-1))/2) > q) --ii;
  while (((ii+1)*(NNODE-1) - ((ii+1)*ii)/2) <= q) ++ii;
  int si = ii*(NNODE-1) - (ii*(ii-1))/2;
  int jj = q - si + ii + 1;

  const float* Ag = Aarr + g*NNODE*EMB;
  const float* Bg = Barr + g*NNODE*EMB;

  // h[m][k] = lrelu(A[i_m,k] + B[j_m,k] + bias[k]) -> LDS in row-major bf16
  #pragma unroll
  for (int m=0; m<16; m++){
    int im = __shfl(ii, m);
    int jm = __shfl(jj, m);
    float v = Ag[im*EMB + lane] + Bg[jm*EMB + lane] + biasv;
    hbuf[w][m][lane] = f2bf(lrelu(v));
  }

  // A operand (16-bit 16x32 layout): row = col, k halves selected by kh
  uint4 alo = *(const uint4*)&hbuf[w][col][kh*8];
  uint4 ahi = *(const uint4*)&hbuf[w][col][16 + kh*8];
  v8u au = { alo.x, alo.y, alo.z, alo.w, ahi.x, ahi.y, ahi.z, ahi.w };
  v16bf a = __builtin_bit_cast(v16bf, au);

  v8f zero = {};
  v8f c0 = __builtin_amdgcn_wmma_f32_16x16x32_bf16(false, a, false, wb0, (short)0, zero, false, false);
  v8f c1 = __builtin_amdgcn_wmma_f32_16x16x32_bf16(false, a, false, wb1, (short)0, zero, false, false);

  // lrelu, transpose D layout back to row-major bf16 via LDS
  #pragma unroll
  for (int v=0; v<8; v++){
    hbuf[w][v + 8*kh][col]      = f2bf(lrelu(c0[v]));
    hbuf[w][v + 8*kh][col + 16] = f2bf(lrelu(c1[v]));
  }

  uint4 blo = *(const uint4*)&hbuf[w][col][kh*8];
  uint4 bhi = *(const uint4*)&hbuf[w][col][16 + kh*8];
  v8u bu = { blo.x, blo.y, blo.z, blo.w, bhi.x, bhi.y, bhi.z, bhi.w };
  v16bf a2 = __builtin_bit_cast(v16bf, bu);

  v8f d3 = __builtin_amdgcn_wmma_f32_16x16x32_bf16(false, a2, false, wb3, (short)0, zero, false, false);

  if (col < NCLS){                              // only cols 0,1 are live
    #pragma unroll
    for (int v=0; v<8; v++){
      int m = v + 8*kh;
      out[(long)(p0 + m)*NCLS + col] = d3[v];
    }
  }
}

// ------------------------------------------------------------------- launch
extern "C" void kernel_launch(void* const* d_in, const int* in_sizes, int n_in,
                              void* d_out, int out_size, void* d_ws, size_t ws_size,
                              hipStream_t stream) {
  (void)in_sizes; (void)n_in; (void)out_size; (void)ws_size;
  const float* feat0  = (const float*)d_in[0];
  const float* feat1  = (const float*)d_in[1];
  const float* ec1_w1 = (const float*)d_in[2];
  const float* ec1_b1 = (const float*)d_in[3];
  const float* ec1_w2 = (const float*)d_in[4];
  const float* ec1_b2 = (const float*)d_in[5];
  const float* ec2_w1 = (const float*)d_in[6];
  const float* ec2_b1 = (const float*)d_in[7];
  const float* ec2_w2 = (const float*)d_in[8];
  const float* ec2_b2 = (const float*)d_in[9];
  const float* bn_g   = (const float*)d_in[10];
  const float* bn_b   = (const float*)d_in[11];
  const float* lin1_w = (const float*)d_in[12];
  const float* lin1_b = (const float*)d_in[13];
  const float* lin2_w = (const float*)d_in[14];
  const float* lin2_b = (const float*)d_in[15];  (void)lin2_b; // zeros in reference init, but keep ABI
  const float* lin3_w = (const float*)d_in[16];
  const float* lin3_b = (const float*)d_in[17];  (void)lin3_b;

  char* ws = (char*)d_ws;
  size_t off = 0;
  auto alloc = [&](size_t bytes)->char*{
    char* p = ws + off;
    off = (off + bytes + 255) & ~(size_t)255;
    return p;
  };
  int*          idxbuf = (int*)         alloc(2*NNODE*K_NB*sizeof(int));
  float*        e1     = (float*)       alloc(2*NNODE*EO*sizeof(float));
  float*        e2     = (float*)       alloc(2*NNODE*EO*sizeof(float));
  float*        Aarr   = (float*)       alloc(2*NNODE*EMB*sizeof(float));
  float*        Barr   = (float*)       alloc(2*NNODE*EMB*sizeof(float));
  float*        scale  = (float*)       alloc(FEAT*sizeof(float));
  float*        shift  = (float*)       alloc(FEAT*sizeof(float));
  float*        biasp  = (float*)       alloc(EMB*sizeof(float));
  unsigned int* w2pack = (unsigned int*)alloc(2*32*8*sizeof(unsigned int));
  unsigned int* w3pack = (unsigned int*)alloc(32*8*sizeof(unsigned int));

  knn_kernel<<<6, 256, 0, stream>>>(feat0, feat1, idxbuf);
  edgeconv_kernel<IN_F><<<192, 256, 0, stream>>>(feat0, feat1,
      ec1_w1, ec1_b1, ec1_w2, ec1_b2, idxbuf, e1);
  edgeconv_kernel<EO><<<192, 256, 0, stream>>>(e1, e1 + NNODE*EO,
      ec2_w1, ec2_b1, ec2_w2, ec2_b2, idxbuf, e2);
  stats_kernel<<<1, 288, 0, stream>>>(feat0, feat1, e1, e2, bn_g, bn_b, scale, shift);
  fold_kernel<<<1, 32, 0, stream>>>(shift, lin1_w, lin1_b, lin2_w, lin3_w,
      biasp, w2pack, w3pack);
  nodeproj_kernel<<<192, 256, 0, stream>>>(feat0, feat1, e1, e2, scale, lin1_w, Aarr, Barr);
  pair_kernel<<<NTILE/8, 256, 0, stream>>>(Aarr, Barr, biasp, w2pack, w3pack, (float*)d_out);
}
